// Attention_42460046688974
// MI455X (gfx1250) — compile-verified, matrix operand order checked by reading, not run on hardware
//
#include <hip/hip_runtime.h>

// Problem constants (match reference)
constexpr int B = 8, N = 1024, S = 1024, C = 768, H = 12, D = 64;
constexpr float SCALE = 0.125f; // 64^-0.5

typedef __attribute__((ext_vector_type(16))) _Float16 v16h;
typedef __attribute__((ext_vector_type(8)))  float    v8f;

// ---------------------------------------------------------------------------
// WMMA helpers
// ---------------------------------------------------------------------------
__device__ __forceinline__ v8f wmma32(v16h a, v16h b, v8f c) {
  // v_wmma_f32_16x16x32_f16 : D = A(16x32) * B(32x16) + C(16x16 f32)
  return __builtin_amdgcn_wmma_f32_16x16x32_f16(
      /*neg_a=*/false, a, /*neg_b=*/false, b,
      /*c_mod=*/(short)0, c, /*reuse_a=*/false, /*reuse_b=*/false);
}

// A/B fragment, 16x32 f16, "row-major" semantics:
//   fragment row = lane & 15, half = lane >> 4
//   element e maps to K = (e>>3)*16 + half*8 + (e&7)
// => per lane: two contiguous 16-byte loads at (row, half*8) and (row, half*8+16)
__device__ __forceinline__ v16h load_frag_f16(const _Float16* tile, int ld) {
  const int lane = threadIdx.x & 31;
  const int r  = lane & 15;
  const int h8 = (lane >> 4) << 3;           // 0 or 8
  const _Float16* p = tile + (size_t)r * ld + h8;
  union { v16h v; uint4 q[2]; } u;
  u.q[0] = *(const uint4*)(p);
  u.q[1] = *(const uint4*)(p + 16);
  return u.v;
}

__device__ __forceinline__ float wave_max(float v) {
  #pragma unroll
  for (int off = 16; off; off >>= 1) v = fmaxf(v, __shfl_xor(v, off, 32));
  return v;
}
__device__ __forceinline__ float wave_sum(float v) {
  #pragma unroll
  for (int off = 16; off; off >>= 1) v += __shfl_xor(v, off, 32);
  return v;
}

// ---------------------------------------------------------------------------
// Kernel 0: f32 -> f16 bulk conversion (8 elements / thread, b128 in/out)
// ---------------------------------------------------------------------------
__global__ __launch_bounds__(256) void cvt_f16_kernel(
    const float* __restrict__ src, _Float16* __restrict__ dst, int n8) {
  const int i = blockIdx.x * blockDim.x + threadIdx.x;
  if (i >= n8) return;
  const float4* s = (const float4*)src + 2 * (size_t)i;
  const float4 a = s[0], b = s[1];
  union { _Float16 h[8]; uint4 q; } u;
  u.h[0] = (_Float16)a.x; u.h[1] = (_Float16)a.y;
  u.h[2] = (_Float16)a.z; u.h[3] = (_Float16)a.w;
  u.h[4] = (_Float16)b.x; u.h[5] = (_Float16)b.y;
  u.h[6] = (_Float16)b.z; u.h[7] = (_Float16)b.w;
  ((uint4*)dst)[i] = u.q;
}

// ---------------------------------------------------------------------------
// Kernel 1: QKV projections  y = x @ W.T  (all-f16 operands, f32 accum)
//   z==0: Q -> qh[b][h][n][d]      z==1: K -> kh[b][h][s][d]
//   z==2: V -> vth[b][h][d][s]  (transposed so attn@V B-frags are row loads)
// block = 128 threads (4 waves); wave tile = 32(M) x 64(Ncols)
// ---------------------------------------------------------------------------
__global__ __launch_bounds__(128) void proj_qkv_kernel(
    const _Float16* __restrict__ xq, const _Float16* __restrict__ xk,
    const _Float16* __restrict__ xv, const _Float16* __restrict__ wq,
    const _Float16* __restrict__ wk, const _Float16* __restrict__ wv,
    _Float16* __restrict__ qh, _Float16* __restrict__ kh,
    _Float16* __restrict__ vth) {
  const int which = blockIdx.z;
  const _Float16* X = which == 0 ? xq : (which == 1 ? xk : xv);
  const _Float16* W = which == 0 ? wq : (which == 1 ? wk : wv);

  const int wave = threadIdx.x >> 5;
  const int lane = threadIdx.x & 31;
  const int m0 = blockIdx.x * 128 + wave * 32;  // rows of [B*seq, C]
  const int n0 = blockIdx.y * 64;               // output channel base

  v8f acc[2][4] = {};
  for (int k = 0; k < C; k += 32) {
    const v16h a0 = load_frag_f16(X + (size_t)m0 * C + k, C);
    const v16h a1 = load_frag_f16(X + (size_t)(m0 + 16) * C + k, C);
    #pragma unroll
    for (int j = 0; j < 4; ++j) {
      // B[k][n] = W[n][k]  -> fragment row = n, contiguous k
      const v16h bf = load_frag_f16(W + (size_t)(n0 + 16 * j) * C + k, C);
      acc[0][j] = wmma32(a0, bf, acc[0][j]);
      acc[1][j] = wmma32(a1, bf, acc[1][j]);
    }
  }

  const int nn = lane & 15;
  const int mh = (lane >> 4) * 8;
  #pragma unroll
  for (int g = 0; g < 2; ++g) {
    #pragma unroll
    for (int j = 0; j < 4; ++j) {
      #pragma unroll
      for (int r = 0; r < 8; ++r) {
        const int m  = m0 + 16 * g + mh + r;  // global row -> (b, t)
        const int nc = n0 + 16 * j + nn;      // channel    -> (h, d)
        const int bb = m >> 10, t = m & 1023;
        const int hh = nc >> 6, d = nc & 63;
        const _Float16 val = (_Float16)acc[g][j][r];
        if (which == 0)      qh [(((size_t)(bb * H + hh)) * N + t) * D + d] = val;
        else if (which == 1) kh [(((size_t)(bb * H + hh)) * S + t) * D + d] = val;
        else                 vth[(((size_t)(bb * H + hh)) * D + d) * S + t] = val;
      }
    }
  }
}

// ---------------------------------------------------------------------------
// Kernel 2: attention core. One block (8 waves) per (b, h, 16-row block).
//   phase 1: scores[16][1024] = scale * Q Kt  (WMMA, waves split S)   -> LDS
//   phase 2: per-row log-softmax, + attn_weight (NT load), write log_attn
//            (NT store), second softmax, attn -> f16 LDS tile
//   phase 3: O tile = attn @ V; all 8 waves (S split in half), LDS-reduce
// dynamic LDS: 16*1024*4 (f32 scores/logits) + 16*1024*2 (f16 attn) = 96 KB
// ---------------------------------------------------------------------------
__global__ __launch_bounds__(256) void attn_kernel(
    const _Float16* __restrict__ qh, const _Float16* __restrict__ kh,
    const _Float16* __restrict__ vth, const float* __restrict__ attnw,
    float* __restrict__ log_attn, _Float16* __restrict__ oh) {
  extern __shared__ char smem[];
  float*     sc = (float*)smem;                       // [16][1024] scores/logits
  _Float16*  at = (_Float16*)(smem + 16 * 1024 * 4);  // [16][1024] attn f16

  int idx = blockIdx.x;
  const int nblk = idx & 63;  idx >>= 6;              // N/16 = 64
  const int h = idx % H;
  const int b = idx / H;
  const int n0 = nblk * 16;
  const int wave = threadIdx.x >> 5;
  const int lane = threadIdx.x & 31;

  const _Float16* Qt = qh  + (((size_t)(b * H + h)) * N + n0) * D;
  const _Float16* Kt = kh  + ((size_t)(b * H + h)) * S * D;
  const _Float16* Vt = vth + ((size_t)(b * H + h)) * D * S;

  // ---- Phase 1: scores tile ----
  const v16h a0 = load_frag_f16(Qt + 0,  D);
  const v16h a1 = load_frag_f16(Qt + 32, D);
  const int nn = lane & 15;
  const int mh = (lane >> 4) * 8;
  #pragma unroll
  for (int j = 0; j < 8; ++j) {
    const int s0 = wave * 128 + j * 16;
    const v16h b0 = load_frag_f16(Kt + (size_t)s0 * D + 0,  D);
    const v16h b1 = load_frag_f16(Kt + (size_t)s0 * D + 32, D);
    v8f c = {};
    c = wmma32(a0, b0, c);
    c = wmma32(a1, b1, c);
    #pragma unroll
    for (int r = 0; r < 8; ++r)
      sc[(mh + r) * 1024 + s0 + nn] = c[r] * SCALE;
  }
  __syncthreads();

  // ---- Phase 2: softmax pipeline, rows 2*wave and 2*wave+1 ----
  for (int rr = 0; rr < 2; ++rr) {
    const int row = wave * 2 + rr;
    float* srow = sc + row * 1024;
    const size_t gidx = (((size_t)b * H + h) * N + (n0 + row)) * S;
    const float* awrow = attnw + gidx;
    float* larow = log_attn + gidx;

    float m1 = -3.0e38f;
    for (int c0 = lane; c0 < S; c0 += 32) m1 = fmaxf(m1, srow[c0]);
    m1 = wave_max(m1);
    float e1 = 0.f;
    for (int c0 = lane; c0 < S; c0 += 32) e1 += __expf(srow[c0] - m1);
    e1 = wave_sum(e1);
    const float logZ = m1 + __logf(e1);

    float m2 = -3.0e38f;
    for (int c0 = lane; c0 < S; c0 += 32) {
      // streaming 403MB in + 403MB out: keep it out of L2 (non-temporal)
      const float l = __builtin_nontemporal_load(awrow + c0) + srow[c0] - logZ;
      srow[c0] = l;
      __builtin_nontemporal_store(l, larow + c0);   // logits output
      m2 = fmaxf(m2, l);
    }
    m2 = wave_max(m2);
    float e2 = 0.f;
    for (int c0 = lane; c0 < S; c0 += 32) e2 += __expf(srow[c0] - m2);
    e2 = wave_sum(e2);
    const float inv = 1.0f / e2;

    _Float16* arow = at + row * 1024;
    for (int c0 = lane; c0 < S; c0 += 32)
      arow[c0] = (_Float16)(__expf(srow[c0] - m2) * inv);
  }
  __syncthreads();

  // ---- Phase 3: O tile = attn @ V, all 8 waves ----
  // waves 0..3 handle s in [0,512), waves 4..7 handle s in [512,1024),
  // each pair covers d-columns [16*(wave&3), +16). Partials reduced via LDS.
  const int dgrp = wave & 3;
  const int d0 = dgrp * 16;
  const int sbase = (wave >> 2) * 512;
  v8f c = {};
  for (int s0 = sbase; s0 < sbase + 512; s0 += 32) {
    const v16h af = load_frag_f16(at + s0, 1024);                 // LDS reads
    const v16h bf = load_frag_f16(Vt + (size_t)d0 * S + s0, S);   // V^T rows
    c = wmma32(af, bf, c);
  }
  float* red = sc;  // score buffer is dead now; 4*16*16 f32 = 4 KB
  if (wave >= 4) {
    #pragma unroll
    for (int r = 0; r < 8; ++r)
      red[dgrp * 256 + (mh + r) * 16 + nn] = c[r];
  }
  __syncthreads();
  if (wave < 4) {
    #pragma unroll
    for (int r = 0; r < 8; ++r) {
      const float v = c[r] + red[dgrp * 256 + (mh + r) * 16 + nn];
      const int m = n0 + mh + r;
      const int d = d0 + nn;
      // store as [b][n][h][d] == row-major [B*N, C] for the output GEMM
      oh[(((size_t)b * N + m) * H + h) * D + d] = (_Float16)v;
    }
  }
}

// ---------------------------------------------------------------------------
// Kernel 3: out = O @ Wo.T + bo   (all-f16 operands, f32 out)
// block = 128 threads (4 waves); wave tile = 32(M) x 64(N)
// ---------------------------------------------------------------------------
__global__ __launch_bounds__(128) void out_proj_kernel(
    const _Float16* __restrict__ oh, const _Float16* __restrict__ wo,
    const float* __restrict__ bo, float* __restrict__ out) {
  const int wave = threadIdx.x >> 5;
  const int lane = threadIdx.x & 31;
  const int m0 = blockIdx.x * 128 + wave * 32;
  const int n0 = blockIdx.y * 64;

  v8f acc[2][4] = {};
  for (int k = 0; k < C; k += 32) {
    const v16h a0 = load_frag_f16(oh + (size_t)m0 * C + k, C);
    const v16h a1 = load_frag_f16(oh + (size_t)(m0 + 16) * C + k, C);
    #pragma unroll
    for (int j = 0; j < 4; ++j) {
      const v16h bf = load_frag_f16(wo + (size_t)(n0 + 16 * j) * C + k, C);
      acc[0][j] = wmma32(a0, bf, acc[0][j]);
      acc[1][j] = wmma32(a1, bf, acc[1][j]);
    }
  }

  const int nn = lane & 15;
  const int mh = (lane >> 4) * 8;
  #pragma unroll
  for (int g = 0; g < 2; ++g) {
    #pragma unroll
    for (int j = 0; j < 4; ++j) {
      const int ncol = n0 + 16 * j + nn;
      const float bias = bo[ncol];
      #pragma unroll
      for (int r = 0; r < 8; ++r)
        out[(size_t)(m0 + 16 * g + mh + r) * C + ncol] = acc[g][j][r] + bias;
    }
  }
}

// ---------------------------------------------------------------------------
// Launch
// ---------------------------------------------------------------------------
extern "C" void kernel_launch(void* const* d_in, const int* in_sizes, int n_in,
                              void* d_out, int out_size, void* d_ws, size_t ws_size,
                              hipStream_t stream) {
  const float* query = (const float*)d_in[0];
  const float* key   = (const float*)d_in[1];
  const float* value = (const float*)d_in[2];
  const float* attnw = (const float*)d_in[3];
  const float* Wq    = (const float*)d_in[4];
  const float* Wk    = (const float*)d_in[5];
  const float* Wv    = (const float*)d_in[6];
  const float* Wo    = (const float*)d_in[7];
  const float* bo    = (const float*)d_in[8];

  float* out      = (float*)d_out;               // [B,N,C]
  float* log_attn = out + (size_t)B * N * C;     // [B,H,N,S]

  const size_t XE = (size_t)B * N * C;           // activation elements (= B*S*C)
  const size_t WE = (size_t)C * C;               // weight elements
  _Float16* qh  = (_Float16*)d_ws;               // [B,H,N,D]
  _Float16* kh  = qh  + (size_t)B * H * N * D;   // [B,H,S,D]
  _Float16* vt  = kh  + (size_t)B * H * S * D;   // [B,H,D,S]
  _Float16* ohb = vt  + (size_t)B * H * D * S;   // [B,N,H,D] == [B*N, C]
  _Float16* xqh = ohb + XE;                      // f16 copies of inputs/weights
  _Float16* xkh = xqh + XE;
  _Float16* xvh = xkh + XE;
  _Float16* wqh = xvh + XE;
  _Float16* wkh = wqh + WE;
  _Float16* wvh = wkh + WE;
  _Float16* woh = wvh + WE;

  // ---- convert f32 operands to f16 once ----
  {
    const int nX8 = (int)(XE / 8), nW8 = (int)(WE / 8);
    const int tb = 256;
    cvt_f16_kernel<<<(nX8 + tb - 1) / tb, tb, 0, stream>>>(query, xqh, nX8);
    cvt_f16_kernel<<<(nX8 + tb - 1) / tb, tb, 0, stream>>>(key,   xkh, nX8);
    cvt_f16_kernel<<<(nX8 + tb - 1) / tb, tb, 0, stream>>>(value, xvh, nX8);
    cvt_f16_kernel<<<(nW8 + tb - 1) / tb, tb, 0, stream>>>(Wq, wqh, nW8);
    cvt_f16_kernel<<<(nW8 + tb - 1) / tb, tb, 0, stream>>>(Wk, wkh, nW8);
    cvt_f16_kernel<<<(nW8 + tb - 1) / tb, tb, 0, stream>>>(Wv, wvh, nW8);
    cvt_f16_kernel<<<(nW8 + tb - 1) / tb, tb, 0, stream>>>(Wo, woh, nW8);
  }

  dim3 gproj(B * N / 128, C / 64, 3);            // 64 x 12 x 3
  proj_qkv_kernel<<<gproj, 128, 0, stream>>>(xqh, xkh, xvh, wqh, wkh, wvh,
                                             qh, kh, vt);

  dim3 gattn(B * H * (N / 16));                  // 6144 blocks
  size_t lds = 16 * 1024 * 4 + 16 * 1024 * 2;    // 96 KB dynamic LDS
  attn_kernel<<<gattn, 256, lds, stream>>>(qh, kh, vt, attnw, log_attn, ohb);

  dim3 gout(B * N / 128, C / 64);                // 64 x 12
  out_proj_kernel<<<gout, 128, 0, stream>>>(ohb, Wo ? woh : woh, bo, out);
}
